// SchNet_20220706030036
// MI455X (gfx1250) — matrix operand order repaired and checked by reading, compiled, not run on hardware
//
#include <hip/hip_runtime.h>
#include <hip/hip_bf16.h>
#include <stdint.h>

typedef __attribute__((ext_vector_type(16))) _Float16 v16h;
typedef __attribute__((ext_vector_type(8)))  float    v8f;
typedef __attribute__((ext_vector_type(4)))  unsigned int u32x4;
typedef __attribute__((ext_vector_type(8)))  int      i32x8;
typedef __attribute__((ext_vector_type(4)))  int      i32x4;

#define FDIM 128
#define GDIM 50

static __device__ __forceinline__ float sspf(float x) {
    // softplus(x) - log(2), numerically stable
    float sp = (x > 20.f) ? x : __logf(1.f + __expf(x));
    return sp - 0.69314718056f;
}

// Dense 16-bit A-matrix layout (ISA 7.12.2): half h of lane (m, hi) holds K:
//   h in [0,8):  K = 8*hi + h
//   h in [8,16): K = 16 + 8*hi + (h-8)
static __device__ __forceinline__ int kmapA(int h, int hi) {
    return (h < 8) ? (8 * hi + h) : (16 + 8 * hi + (h - 8));
}

// ---------------- TDM: DMA n8 x 8-byte elements from global into LDS ----------------
// D# per ISA 8.3/8.4: 1-D tile, data_size=8B, tensor_dim0=tile_dim0=n8, dim1=1.
static __device__ __forceinline__ void tdm_load_to_lds(const void* gsrc,
                                                       unsigned int lds_off,
                                                       unsigned int n8) {
    unsigned long long ga = (unsigned long long)(uintptr_t)gsrc;
    u32x4 g0;
    g0[0] = 1u;                                            // count=1, user mode
    g0[1] = lds_off;                                       // lds_addr (bytes)
    g0[2] = (unsigned int)ga;                              // global_addr[31:0]
    g0[3] = (unsigned int)((ga >> 32) & 0x01FFFFFFu)       // global_addr[56:32]
            | (2u << 30);                                  // type=2 ("image")
    i32x8 g1;
    g1[0] = (int)(3u << 16);                               // data_size=3 (8B)
    g1[1] = (int)((n8 & 0xFFFFu) << 16);                   // tensor_dim0[15:0]
    g1[2] = (int)(((n8 >> 16) & 0xFFFFu) | (1u << 16));    // tensor_dim0[31:16], tensor_dim1=1
    g1[3] = (int)((n8 & 0xFFFFu) << 16);                   // tile_dim0
    g1[4] = 0;                                             // tile_dim1/2 unused
    g1[5] = (int)n8;                                       // tensor_dim0_stride
    g1[6] = 0;
    g1[7] = 0;
    i32x4 z4 = {0, 0, 0, 0};
#if __clang_major__ >= 23
    i32x8 z8 = {0, 0, 0, 0, 0, 0, 0, 0};
    __builtin_amdgcn_tensor_load_to_lds(g0, g1, z4, z4, z8, 0);
#else
    __builtin_amdgcn_tensor_load_to_lds(g0, g1, z4, z4, 0);
#endif
}

// ---------------- weight packing (f32 row-major [rows,cols] -> WMMA-B f16 tiles) ----
// B 32x16 f16 layout (matches dense A K order): lane = (n%16) + 16*hi,
// Tile (kt, nt): k = kt*32 + kmapA(h, hi), n = nt*16 + (lane&15).
__global__ void pack_b_kernel(const float* __restrict__ src, _Float16* __restrict__ dst,
                              int rows, int cols, int KT, int NT) {
    int idx = blockIdx.x * blockDim.x + threadIdx.x;
    int total = KT * NT * 512;
    if (idx >= total) return;
    int tile = idx >> 9;
    int lane = (idx >> 4) & 31;
    int h = idx & 15;
    int kt = tile / NT, nt = tile % NT;
    int hi = lane >> 4;
    int n = nt * 16 + (lane & 15);
    int k = kt * 32 + kmapA(h, hi);
    float v = (k < rows && n < cols) ? src[(size_t)k * cols + n] : 0.f;
    dst[idx] = (_Float16)v;
}

static __device__ __forceinline__ v16h loadB_packed(const _Float16* __restrict__ pk,
                                                    int tile, int lane) {
    return *(const v16h*)(pk + ((size_t)tile * 32 + lane) * 16);
}

// Load A 16x32 f16 chunk from row-major f32 [*, FDIM]; optional ssp on input.
static __device__ __forceinline__ v16h loadA_global(const float* __restrict__ base,
                                                    int row, int k0, int hi, bool do_ssp) {
    const float* p = base + (size_t)row * FDIM + k0 + 8 * hi;
    float4 q0 = *(const float4*)(p);
    float4 q1 = *(const float4*)(p + 4);
    float4 q2 = *(const float4*)(p + 16);
    float4 q3 = *(const float4*)(p + 20);
    float t[16] = {q0.x,q0.y,q0.z,q0.w, q1.x,q1.y,q1.z,q1.w,
                   q2.x,q2.y,q2.z,q2.w, q3.x,q3.y,q3.z,q3.w};
    v16h a;
#pragma unroll
    for (int h = 0; h < 16; ++h) {
        float v = t[h];
        if (do_ssp) v = sspf(v);
        a[h] = (_Float16)v;
    }
    return a;
}

// ---------------- misc small kernels ----------------
__global__ void zero_kernel(float* p, long n) {
    long i = (long)blockIdx.x * blockDim.x + threadIdx.x;
    if (i < n) p[i] = 0.f;
}

__global__ void init_h_kernel(const float* __restrict__ emb, const int* __restrict__ z,
                              float* __restrict__ h, int N) {
    long i = (long)blockIdx.x * blockDim.x + threadIdx.x;
    if (i >= (long)N * FDIM) return;
    int n = (int)(i >> 7), f = (int)(i & 127);
    h[i] = emb[(size_t)z[n] * FDIM + f];
}

// ---------------- edge filter: W = (ssp(gauss(ew)@mw1+mb1)@mw2+mb2)*C(ew) -> f16 ----
__global__ void __launch_bounds__(128)
edge_filter_kernel(const float* __restrict__ ew_g,
                   const _Float16* __restrict__ mw1p,   // 2x8 tiles
                   const _Float16* __restrict__ mw2p,   // 4x8 tiles
                   const float* __restrict__ mb1,
                   const float* __restrict__ mb2,
                   _Float16* __restrict__ Wout,
                   int E) {
    __shared__ float sH[4][16][132];   // per-wave 16x128 (+pad 4) transpose buffer
    __shared__ float sEw[4][16];
    const int wave = threadIdx.x >> 5;
    const int lane = threadIdx.x & 31;
    const int m = lane & 15, hi = lane >> 4;
    const int ntiles = (E + 15) / 16;
    int tile = blockIdx.x * 4 + wave;
    const bool valid = (tile < ntiles);
    if (!valid) tile = ntiles - 1;
    const int e0 = tile * 16;

    const float DELTA = 6.0f / 49.0f;
    const float COEF  = -0.5f / (DELTA * DELTA);

    float ew_own = ew_g[min(e0 + m, E - 1)];
    if (hi == 0) sEw[wave][m] = ew_own;

    // ---- stage 1: [16,64(pad from 50)] x [64,128] ----
    v8f acc1[8] = {};
#pragma unroll
    for (int kc = 0; kc < 2; ++kc) {
        v16h a;
#pragma unroll
        for (int h = 0; h < 16; ++h) {
            int k = kc * 32 + kmapA(h, hi);
            float v = 0.f;
            if (k < GDIM) { float d = ew_own - (float)k * DELTA; v = __expf(COEF * d * d); }
            a[h] = (_Float16)v;
        }
#pragma unroll
        for (int nt = 0; nt < 8; ++nt) {
            v16h b = loadB_packed(mw1p, kc * 8 + nt, lane);
            acc1[nt] = __builtin_amdgcn_wmma_f32_16x16x32_f16(
                false, a, false, b, (short)0, acc1[nt], false, false);
        }
    }
    // bias + ssp, C-layout -> LDS row-major [16][128]
#pragma unroll
    for (int nt = 0; nt < 8; ++nt) {
        int n = nt * 16 + m;
        float b1 = mb1[n];
#pragma unroll
        for (int r = 0; r < 8; ++r)
            sH[wave][r + 8 * hi][n] = sspf(acc1[nt][r] + b1);
    }
    __syncthreads();

    // ---- stage 2: [16,128] x [128,128] ----
    v8f acc2[8] = {};
#pragma unroll
    for (int kc = 0; kc < 4; ++kc) {
        v16h a;
        const float* rowp = &sH[wave][m][kc * 32 + 8 * hi];
#pragma unroll
        for (int h = 0; h < 8; ++h) a[h] = (_Float16)rowp[h];
#pragma unroll
        for (int h = 0; h < 8; ++h) a[8 + h] = (_Float16)rowp[16 + h];
#pragma unroll
        for (int nt = 0; nt < 8; ++nt) {
            v16h b = loadB_packed(mw2p, kc * 8 + nt, lane);
            acc2[nt] = __builtin_amdgcn_wmma_f32_16x16x32_f16(
                false, a, false, b, (short)0, acc2[nt], false, false);
        }
    }
    // bias + cosine cutoff envelope, store f16 row-major [E,128]
#pragma unroll
    for (int nt = 0; nt < 8; ++nt) {
        int n = nt * 16 + m;
        float b2 = mb2[n];
#pragma unroll
        for (int r = 0; r < 8; ++r) {
            int rowe = r + 8 * hi;
            int e = e0 + rowe;
            float ew = sEw[wave][rowe];
            float cenv = 0.5f * (__cosf(ew * 0.52359877559f) + 1.0f);  // pi/6
            float v = (acc2[nt][r] + b2) * cenv;
            if (valid && e < E) Wout[(size_t)e * FDIM + n] = (_Float16)v;
        }
    }
}

// ---------------- generic node GEMM: out = [ssp?](A) @ Bpack + bias [+ out] ---------
// Packed B (4x8 tiles = 32KB) is DMA'd into LDS once per block via the TDM.
__global__ void __launch_bounds__(256)
node_gemm_kernel(const float* __restrict__ A,
                 const _Float16* __restrict__ Bp,   // 4x8 tiles (16384 halfs)
                 const float* __restrict__ bias,    // nullable
                 float* __restrict__ out,
                 int nrows, int flags) {            // bit0: ssp(A), bit1: out += result
    __shared__ _Float16 sB[16384];                  // 32KB packed B
    if (threadIdx.x < 32) {                         // wave 0 issues the tensor DMA
        tdm_load_to_lds(Bp, (unsigned int)(uintptr_t)&sB[0], 4096u);
        __builtin_amdgcn_s_wait_tensorcnt(0);
    }
    __syncthreads();

    const int wave = threadIdx.x >> 5, lane = threadIdx.x & 31;
    const int m = lane & 15, hi = lane >> 4;
    const int ntiles = (nrows + 15) / 16;
    int tile = blockIdx.x * 8 + wave;
    const bool valid = (tile < ntiles);
    if (!valid) tile = ntiles - 1;
    const int r0 = tile * 16;
    const bool do_ssp = (flags & 1) != 0;

    v8f acc[8] = {};
#pragma unroll
    for (int kc = 0; kc < 4; ++kc) {
        int row = min(r0 + m, nrows - 1);
        v16h a = loadA_global(A, row, kc * 32, hi, do_ssp);
#pragma unroll
        for (int nt = 0; nt < 8; ++nt) {
            v16h b = *(const v16h*)(sB + ((kc * 8 + nt) * 32 + lane) * 16);
            acc[nt] = __builtin_amdgcn_wmma_f32_16x16x32_f16(
                false, a, false, b, (short)0, acc[nt], false, false);
        }
    }
#pragma unroll
    for (int nt = 0; nt < 8; ++nt) {
        int n = nt * 16 + m;
        float bv = bias ? bias[n] : 0.f;
#pragma unroll
        for (int r = 0; r < 8; ++r) {
            int row = r0 + r + 8 * hi;
            if (valid && row < nrows) {
                size_t o = (size_t)row * FDIM + n;
                float v = acc[nt][r] + bv;
                if (flags & 2) v += out[o];
                out[o] = v;
            }
        }
    }
}

// ---------------- gather-modulate-scatter: msum[dst] += xj[src] * W[e] --------------
__global__ void scatter_kernel(const float* __restrict__ xj, const _Float16* __restrict__ W,
                               const int* __restrict__ src, const int* __restrict__ dst,
                               float* __restrict__ msum, int E) {
    long idx = (long)blockIdx.x * blockDim.x + threadIdx.x;
    if (idx >= (long)E * 32) return;
    int e = (int)(idx >> 5);
    int f = (int)(idx & 31) * 4;
    int s = src[e], d = dst[e];
    float4 x = *(const float4*)(xj + (size_t)s * FDIM + f);
    const _Float16* wp = W + (size_t)e * FDIM + f;
    float* mp = msum + (size_t)d * FDIM + f;
    atomicAdd(mp + 0, x.x * (float)wp[0]);
    atomicAdd(mp + 1, x.y * (float)wp[1]);
    atomicAdd(mp + 2, x.z * (float)wp[2]);
    atomicAdd(mp + 3, x.w * (float)wp[3]);
}

// ---------------- readout: o = ssp(h@l1w+l1b)@l2w + l2b; segment-sum over batch -----
__global__ void __launch_bounds__(256)
readout_kernel(const float* __restrict__ h,
               const _Float16* __restrict__ l1wp,   // 4x4 tiles (8192 halfs)
               const float* __restrict__ l1b,
               const float* __restrict__ l2w,
               const float* __restrict__ l2b,
               const int* __restrict__ z,
               const int* __restrict__ batch,
               float* __restrict__ out,
               int N) {
    __shared__ _Float16 sB[8192];                   // 16KB packed l1w
    if (threadIdx.x < 32) {
        tdm_load_to_lds(l1wp, (unsigned int)(uintptr_t)&sB[0], 2048u);
        __builtin_amdgcn_s_wait_tensorcnt(0);
    }
    __syncthreads();

    const int wave = threadIdx.x >> 5, lane = threadIdx.x & 31;
    const int m = lane & 15, hi = lane >> 4;
    const int ntiles = (N + 15) / 16;
    int tile = blockIdx.x * 8 + wave;
    const bool valid = (tile < ntiles);
    if (!valid) tile = ntiles - 1;
    const int r0 = tile * 16;

    v8f acc[4] = {};
#pragma unroll
    for (int kc = 0; kc < 4; ++kc) {
        int row = min(r0 + m, N - 1);
        v16h a = loadA_global(h, row, kc * 32, hi, false);
#pragma unroll
        for (int nt = 0; nt < 4; ++nt) {
            v16h b = *(const v16h*)(sB + ((kc * 4 + nt) * 32 + lane) * 16);
            acc[nt] = __builtin_amdgcn_wmma_f32_16x16x32_f16(
                false, a, false, b, (short)0, acc[nt], false, false);
        }
    }
    float p[8] = {};
#pragma unroll
    for (int nt = 0; nt < 4; ++nt) {
        int n = nt * 16 + m;
        float c1 = l1b[n];
        float w2 = l2w[n];
#pragma unroll
        for (int r = 0; r < 8; ++r) p[r] += sspf(acc[nt][r] + c1) * w2;
    }
    // reduce across the 16 column-lanes of each half
#pragma unroll
    for (int s = 1; s < 16; s <<= 1) {
#pragma unroll
        for (int r = 0; r < 8; ++r) p[r] += __shfl_xor(p[r], s, 32);
    }
    if (valid && m == 0) {  // lanes 0 (rows 0..7) and 16 (rows 8..15)
        float b2 = l2b[0];
#pragma unroll
        for (int r = 0; r < 8; ++r) {
            int row = r0 + r + 8 * hi;
            if (row < N && z[row] != 0)
                atomicAdd(&out[batch[row]], p[r] + b2);
        }
    }
}

// ====================================================================================
extern "C" void kernel_launch(void* const* d_in, const int* in_sizes, int n_in,
                              void* d_out, int out_size, void* d_ws, size_t ws_size,
                              hipStream_t stream) {
    const float* edge_weight = (const float*)d_in[0];
    const float* emb   = (const float*)d_in[1];
    const float* mw1   = (const float*)d_in[2];
    const float* mb1   = (const float*)d_in[3];
    const float* mw2   = (const float*)d_in[4];
    const float* mb2   = (const float*)d_in[5];
    const float* cf1   = (const float*)d_in[6];
    const float* cf2w  = (const float*)d_in[7];
    const float* cf2b  = (const float*)d_in[8];
    const float* blw   = (const float*)d_in[9];
    const float* blb   = (const float*)d_in[10];
    const float* l1w   = (const float*)d_in[11];
    const float* l1b   = (const float*)d_in[12];
    const float* l2w   = (const float*)d_in[13];
    const float* l2b   = (const float*)d_in[14];
    const int*   z     = (const int*)d_in[15];
    const int*   eidx  = (const int*)d_in[16];
    const int*   batch = (const int*)d_in[17];
    float* out = (float*)d_out;

    const int E = in_sizes[0];
    const int N = in_sizes[15];
    const int* src = eidx;
    const int* dst = eidx + E;

    // ---- workspace layout ----
    char* ws = (char*)d_ws;
    const size_t NF4 = (size_t)N * FDIM * 4;
    float*    hbuf = (float*)(ws);
    float*    xj   = (float*)(ws + NF4);          // also reused as tmp for cf2 output
    float*    msum = (float*)(ws + 2 * NF4);
    _Float16* Wbuf = (_Float16*)(ws + 3 * NF4);
    _Float16* packs = (_Float16*)(ws + 3 * NF4 + (size_t)E * FDIM * 2);
    const size_t PL = 8192 + 4 * 16384;           // per-layer packed halfs
    // per layer: mw1 @+0 (2x8), mw2 @+8192 (4x8), cf1 @+24576, cf2w @+40960, blw @+57344
    // l1w @ 2*PL (4x4)

    auto packW = [&](const float* s, _Float16* dp, int rows, int cols, int KT, int NT) {
        int total = KT * NT * 512;
        pack_b_kernel<<<(total + 255) / 256, 256, 0, stream>>>(s, dp, rows, cols, KT, NT);
    };
    for (int l = 0; l < 2; ++l) {
        packW(mw1  + (size_t)l * GDIM * FDIM, packs + l * PL +     0, GDIM, FDIM, 2, 8);
        packW(mw2  + (size_t)l * FDIM * FDIM, packs + l * PL +  8192, FDIM, FDIM, 4, 8);
        packW(cf1  + (size_t)l * FDIM * FDIM, packs + l * PL + 24576, FDIM, FDIM, 4, 8);
        packW(cf2w + (size_t)l * FDIM * FDIM, packs + l * PL + 40960, FDIM, FDIM, 4, 8);
        packW(blw  + (size_t)l * FDIM * FDIM, packs + l * PL + 57344, FDIM, FDIM, 4, 8);
    }
    packW(l1w, packs + 2 * PL, FDIM, 64, 4, 4);

    // zero output (atomically accumulated)
    zero_kernel<<<(out_size + 255) / 256, 256, 0, stream>>>(out, (long)out_size);

    // h = emb[z]
    {
        long nf = (long)N * FDIM;
        init_h_kernel<<<(int)((nf + 255) / 256), 256, 0, stream>>>(emb, z, hbuf, N);
    }

    const int etiles  = (E + 15) / 16;
    const int ntile   = (N + 15) / 16;
    const int nblocks = (ntile + 7) / 8;

    for (int l = 0; l < 2; ++l) {
        _Float16* pk = packs + l * PL;
        // per-edge continuous filter W (f16)
        edge_filter_kernel<<<(etiles + 3) / 4, 128, 0, stream>>>(
            edge_weight, pk, pk + 8192, mb1 + (size_t)l * FDIM, mb2 + (size_t)l * FDIM,
            Wbuf, E);
        // xj = h @ cf1
        node_gemm_kernel<<<nblocks, 256, 0, stream>>>(hbuf, pk + 24576, nullptr, xj, N, 0);
        // msum = scatter_add(xj[src] * W, dst)
        {
            long nf = (long)N * FDIM;
            zero_kernel<<<(int)((nf + 255) / 256), 256, 0, stream>>>(msum, nf);
            long sct = (long)E * 32;
            scatter_kernel<<<(int)((sct + 255) / 256), 256, 0, stream>>>(
                xj, Wbuf, src, dst, msum, E);
        }
        // tmp = msum @ cf2w + cf2b   (xj buffer reused as tmp)
        node_gemm_kernel<<<nblocks, 256, 0, stream>>>(
            msum, pk + 40960, cf2b + (size_t)l * FDIM, xj, N, 0);
        // h += ssp(tmp) @ blw + blb
        node_gemm_kernel<<<nblocks, 256, 0, stream>>>(
            xj, pk + 57344, blb + (size_t)l * FDIM, hbuf, N, 3);
    }

    readout_kernel<<<nblocks, 256, 0, stream>>>(
        hbuf, packs + 2 * PL, l1b, l2w, l2b, z, batch, out, N);
}
